// Block_9199819948105
// MI455X (gfx1250) — matrix operand order repaired and checked by reading, not compiled, not run on hardware
//
#include <hip/hip_runtime.h>

// ---------------- CDNA5 WMMA types ----------------
typedef __attribute__((ext_vector_type(16))) __bf16 v16bf;
typedef __attribute__((ext_vector_type(2)))  __bf16 v2bf;
typedef __attribute__((ext_vector_type(8)))  float  v8f;

// ---------------- problem constants ----------------
#define NB     3
#define KS     7
#define TAPS   343          // 7*7*7
#define BATCH  16
#define CIN    4
#define VEC    8
#define COUT   16
#define DIN    64
#define D1     34           // conv1 output spatial
#define D2     19           // conv2 output spatial
#define NVOX1  (D1*D1*D1)   // 39304
#define NVOX2  (D2*D2*D2)   // 6859
#define C1     24           // conv1 out channels (8 vectors * 3)
#define CMID   96           // conv2 in channels (24 vec + 72 tensor)
#define K1ROWS (49*32)      // conv1 GEMM K padded: 49 (dz,dy) chunks * 32
#define K2ROWS (TAPS*CMID)  // conv2 GEMM K = 32928 = 1029 * 32 exactly

// ---------------- workspace layout (bytes) ----------------
// all offsets multiples of 32 so v16bf loads are 32B-aligned
static const size_t OFF_K1B   = 0;                               // 1568*32 bf16   = 100,352 B
static const size_t OFF_K2B   = 100352;                          // 32928*16 bf16  = 1,053,696 B
static const size_t OFF_VT    = 1154048;                         // 16*39304*96 bf16 = 120,741,888 B
static const size_t OFF_Y     = 121895936;                       // 16*16*6859 f32 = 7,023,616 B
static const size_t OFF_STATS = 128919552;                       // 32 f32
// total required ws: ~128.92 MB

// =====================================================================
// Build conv1 weight table in WMMA-B lane layout: K1B[chunk][L][nt][s] holds
// n = nt*16+(L&15), local k = table(L>>4, s), k = (dz*7+dy)*32 + (dx*4+cin). Entries with local k>=28
// or n>=24 are ZERO -> A-fragment garbage in those K/N slots is harmless.
// =====================================================================
__global__ void build_k1(const float* __restrict__ W1,
                         const float* __restrict__ basis1,
                         __bf16* __restrict__ K1B)
{
    int e = blockIdx.x * blockDim.x + threadIdx.x;
    if (e >= K1ROWS * 32) return;
    int L  = (e >> 5) & 31;
    int sl = e & 15;
    int n  = (e & 16) + (L & 15);
    int chunk = e >> 10;         // 0..48 == dz*7+dy
    int j     = (sl & 8) * 2 + (L >> 4) * 8 + (sl & 7);
    float val = 0.0f;
    if (j < 28 && n < C1) {
        int dx  = j >> 2;
        int cin = j & 3;
        int tap = chunk * KS + dx;
        int u = n / 3, i = n % 3;
        #pragma unroll
        for (int bb = 0; bb < NB; ++bb)
            val += W1[(u * CIN + cin) * NB + bb] * basis1[(bb * 3 + i) * TAPS + tap];
    }
    K1B[e] = (__bf16)val;
}

// =====================================================================
// Build conv2 weight table: K2B[chunk][L][s]: n = L&15, k = chunk*32 + table(L>>4, s), k = tap*96 + cin.
// cin<24: vector part (W2a,basis2a); cin>=24: tensor part (W2b,basis2b).
// =====================================================================
__global__ void build_k2(const float* __restrict__ W2a,
                         const float* __restrict__ W2b,
                         const float* __restrict__ basis2a,
                         const float* __restrict__ basis2b,
                         __bf16* __restrict__ K2B)
{
    int e = blockIdx.x * blockDim.x + threadIdx.x;
    if (e >= K2ROWS * COUT) return;
    int L  = (e >> 4) & 31;
    int sl = e & 15;
    int n  = L & 15;
    int kk = (e >> 9) * 32 + (sl & 8) * 2 + (L >> 4) * 8 + (sl & 7);
    int cin = kk % CMID;
    int tap = kk / CMID;
    float val = 0.0f;
    if (cin < C1) {
        int vch = cin / 3, i = cin % 3;
        #pragma unroll
        for (int bb = 0; bb < NB; ++bb)
            val += W2a[(n * VEC + vch) * NB + bb] * basis2a[(bb * 3 + i) * TAPS + tap];
    } else {
        int c = cin - C1;
        int vch = c / 9, jj = c % 9;
        #pragma unroll
        for (int bb = 0; bb < NB; ++bb)
            val += W2b[(n * VEC + vch) * NB + bb] * basis2b[(bb * 9 + jj) * TAPS + tap];
    }
    K2B[e] = (__bf16)val;
}

// =====================================================================
// conv1 (implicit GEMM, bf16 WMMA, f32 acc) fused with tensor product.
// Block = (b, oz, oy): M = ox (34 pad 48 -> 3 tiles), N = ch (24 pad 32
// -> 2 tiles); 6 waves, one (Mtile,Ntile) each. Double-buffered LDS:
// next chunk's global loads are issued before the current WMMA, and
// converted/stored to the other buffer after it (latency hidden).
// =====================================================================
__global__ __launch_bounds__(192) void conv1_tp_kernel(
        const float* __restrict__ s,
        const __bf16* __restrict__ K1B,
        __bf16* __restrict__ vt)
{
    __shared__ __bf16 sA[2][73 * 4];    // [buf][xoff 0..72][cin]
    __shared__ float  vLDS[32 * 48];    // [ch][m]

    const int bid = blockIdx.x;
    const int b  = bid / (D1 * D1);
    const int rz = bid % (D1 * D1);
    const int oz = rz / D1;
    const int oy = rz % D1;

    const int tid  = threadIdx.x;
    const int lane = tid & 31;
    const int w    = tid >> 5;              // 0..5
    const int mt   = w >> 1;                // M tile 0..2
    const int nt   = w & 1;                 // N tile 0..1
    const int hi   = lane >> 4;
    const int mrow = mt * 16 + (lane & 15); // ox for this lane's A row

    // each thread stages <=2 of the 292 strip entries
    const int cin0 = tid / 73,         xo0 = tid - cin0 * 73;
    const int t1   = tid + 192;
    const int cin1 = t1 / 73,          xo1 = t1 - cin1 * 73;
    const bool has1 = t1 < CIN * 73;

    float rv0 = 0.0f, rv1 = 0.0f;
    auto loadRegs = [&](int idx) {
        const int dz = idx / KS, dy = idx - dz * KS;
        const int iz = oz * 2 - 5 + dz;
        const int iy = oy * 2 - 5 + dy;
        const bool zy = ((unsigned)iz < DIN) && ((unsigned)iy < DIN);
        const float* row0 = s + (((size_t)(b * CIN + cin0) * DIN + iz) * DIN + iy) * DIN;
        const float* row1 = s + (((size_t)(b * CIN + cin1) * DIN + iz) * DIN + iy) * DIN;
        rv0 = 0.0f; rv1 = 0.0f;
        if (zy && (unsigned)(xo0 - 5) < DIN) rv0 = row0[xo0 - 5];
        if (has1 && zy && (unsigned)(xo1 - 5) < DIN) rv1 = row1[xo1 - 5];
    };
    auto storeLDS = [&](int bufn) {
        sA[bufn][xo0 * 4 + cin0] = (__bf16)rv0;
        if (has1) sA[bufn][xo1 * 4 + cin1] = (__bf16)rv1;
    };

    v8f acc = {0.f,0.f,0.f,0.f,0.f,0.f,0.f,0.f};

    loadRegs(0);
    storeLDS(0);
    __syncthreads();

    // pipelined B fragment: lane = N column (L&15) + k half (L>>4), 16 contiguous bf16 (32B load)
    v16bf fb = *(const v16bf*)(K1B + (size_t)lane * 32 + nt * 16);

    for (int chunk = 0; chunk < 49; ++chunk) {
        const int buf = chunk & 1;
        if (chunk < 48) loadRegs(chunk + 1);         // global loads in flight
        v16bf fbn = fb;
        if (chunk < 48)
            fbn = *(const v16bf*)(K1B + ((size_t)(chunk + 1) * 32 + lane) * 32 + nt * 16);

        // ---- A fragment: 16-bit A 16x32 layout,
        //      k0(i,lane) = (i>>2)*16 + (lane>>4)*8 + (i&3)*2 ----
        const __bf16* sa = sA[buf];
        v16bf fa;
        #pragma unroll
        for (int i = 0; i < 8; ++i) {
            const int j0  = (i >> 2) * 16 + hi * 8 + (i & 3) * 2; // even
            const int dx  = j0 >> 2;
            const int cin = j0 & 3;          // 0 or 2 -> pair contiguous
            int xo = mrow * 2 + dx;
            xo = xo < 72 ? xo : 72;          // clamp: padded slots harmless
            const v2bf pr = *(const v2bf*)(sa + xo * 4 + cin);
            fa[2 * i]     = pr[0];
            fa[2 * i + 1] = pr[1];
        }
        acc = __builtin_amdgcn_wmma_f32_16x16x32_bf16(
                  false, fa, false, fb, (short)0, acc, false, false);
        if (chunk < 48) storeLDS(buf ^ 1);           // convert+store after WMMA issue
        fb = fbn;
        __syncthreads();
    }

    // ---- spill D (lane = N, VGPR r -> M = r + 8*hi) ----
    #pragma unroll
    for (int r = 0; r < 8; ++r) {
        const int ch = nt * 16 + (lane & 15);
        const int m  = mt * 16 + r + hi * 8;
        vLDS[ch * 48 + m] = acc[r];
    }
    __syncthreads();

    // ---- tensor product + write vt[b][z][y][x][96] (channel-fastest) ----
    for (int t = tid; t < D1 * VEC; t += 192) {
        const int ox  = t >> 3;
        const int vch = t & 7;
        const float vx = vLDS[(vch * 3 + 0) * 48 + ox];
        const float vy = vLDS[(vch * 3 + 1) * 48 + ox];
        const float vz = vLDS[(vch * 3 + 2) * 48 + ox];
        __bf16* p = vt + ((size_t)b * NVOX1 + (size_t)oz * (D1 * D1) + (size_t)oy * D1 + ox) * CMID;
        p[vch * 3 + 0] = (__bf16)vx;
        p[vch * 3 + 1] = (__bf16)vy;
        p[vch * 3 + 2] = (__bf16)vz;
        __bf16* q = p + C1 + vch * 9;
        const float vv[3] = {vx, vy, vz};
        #pragma unroll
        for (int i = 0; i < 3; ++i)
            #pragma unroll
            for (int j = 0; j < 3; ++j)
                q[i * 3 + j] = (__bf16)(vv[i] * vv[j]);
    }
}

// =====================================================================
// conv2 (implicit GEMM, bf16 WMMA, f32 acc).
// Block = (b, oz, oy): M = ox (19 pad 32 -> 2 tiles), N = 16 ch.
// K-order k = tap*96+cin -> chunk id c has B rows at 32*c+lane (exact).
// 8 waves split {2 M-tiles} x {4 K-groups}; B software-pipelined.
// DOUBLE-BUFFERED async staging: GLOBAL_LOAD_ASYNC_TO_LDS_B128 for the
// next (dz,dy) row strip is issued before the chunk loop and only waited
// (s_wait_asynccnt 0) at iteration end -> DMA overlaps the WMMAs.
// =====================================================================
__global__ __launch_bounds__(256) void conv2_kernel(
        const __bf16* __restrict__ vt,
        const __bf16* __restrict__ K2B,
        float* __restrict__ y)
{
    __shared__ __bf16 sV[2][43 * 96];  // [buf][xoff 0..42][cin]
    __shared__ float  pLDS[8 * 256];   // per-wave partials [w][ch][m_local]

    const int bid = blockIdx.x;
    const int b  = bid / (D2 * D2);
    const int rz = bid % (D2 * D2);
    const int oz = rz / D2;
    const int oy = rz % D2;

    const int tid  = threadIdx.x;
    const int lane = tid & 31;
    const int w    = tid >> 5;              // 0..7
    const int mt   = w & 1;                 // M tile
    const int kg   = w >> 1;                // K group 0..3
    const int hi   = lane >> 4;
    const int mrow = mt * 16 + (lane & 15); // ox

    v8f acc = {0.f,0.f,0.f,0.f,0.f,0.f,0.f,0.f};

    // issue async stage of row strip idx (= dz*7+dy) into buffer bufn
    auto stage = [&](int idx, int bufn) {
        const int dz = idx / KS, dy = idx - dz * KS;
        const int iz = oz * 2 - 5 + dz;
        const int iy = oy * 2 - 5 + dy;
        const bool zyok = ((unsigned)iz < D1) && ((unsigned)iy < D1);
        const __bf16* src = vt + ((size_t)b * NVOX1 + (size_t)iz * (D1 * D1) + (size_t)iy * D1) * CMID;
        for (int t = tid; t < 43 * 12; t += 256) {
            const int xo = t / 12;
            const int cg = t - xo * 12;          // channel group (8 ch)
            const int ix = xo - 5;
            __bf16* dst = &sV[bufn][xo * CMID + cg * 8];     // 16B aligned
            if (zyok && (unsigned)ix < D1) {
                const int voff = (ix * CMID + cg * 8) * 2;   // byte offset
                const unsigned ldsa = (unsigned)(uintptr_t)dst;
                asm volatile("global_load_async_to_lds_b128 %0, %1, %2"
                             :: "v"(ldsa), "v"(voff), "s"(src) : "memory");
            } else {
                unsigned* z = (unsigned*)dst;
                z[0] = 0u; z[1] = 0u; z[2] = 0u; z[3] = 0u;
            }
        }
    };

    stage(0, 0);
    asm volatile("s_wait_asynccnt 0" ::: "memory");
    __syncthreads();

    for (int idx = 0; idx < 49; ++idx) {          // idx = dz*7+dy
        const int buf = idx & 1;
        if (idx < 48) stage(idx + 1, buf ^ 1);    // async DMA overlaps compute

        const __bf16* sb = sV[buf];
        // ---- this wave's K-chunks: c ≡ kg (mod 4) in [base3, base3+21) ----
        const int tapbase = idx * KS;
        const int base3   = tapbase * 3;
        const int cend    = base3 + 21;
        int c = base3 + ((kg - base3) & 3);
        if (c < cend) {
            __builtin_prefetch(K2B + (size_t)c * 32 * COUT, 0, 0);
            v16bf fb = *(const v16bf*)(K2B + ((size_t)c * 32 + lane) * COUT);
            while (true) {
                const int cn = c + 4;
                const bool more = (cn < cend);   // wave-uniform
                v16bf fbn = fb;
                if (more)
                    fbn = *(const v16bf*)(K2B + ((size_t)cn * 32 + lane) * COUT);
                const int tap = c / 3;
                const int cc  = c - tap * 3;
                const int dx  = tap - tapbase;
                int xo = mrow * 2 + dx;
                xo = xo < 42 ? xo : 42;          // clamp: M-pad rows unread
                v16bf fa;
                #pragma unroll
                for (int i = 0; i < 8; ++i) {
                    const int j0  = (i >> 2) * 16 + hi * 8 + (i & 3) * 2;
                    const int cin = cc * 32 + j0;            // even -> 4B pair
                    const v2bf pr = *(const v2bf*)(sb + xo * CMID + cin);
                    fa[2 * i]     = pr[0];
                    fa[2 * i + 1] = pr[1];
                }
                acc = __builtin_amdgcn_wmma_f32_16x16x32_bf16(
                          false, fa, false, fb, (short)0, acc, false, false);
                if (!more) break;
                c = cn; fb = fbn;
            }
        }
        asm volatile("s_wait_asynccnt 0" ::: "memory");
        __syncthreads();
    }

    // ---- deterministic cross-wave reduction via LDS ----
    #pragma unroll
    for (int r = 0; r < 8; ++r)
        pLDS[w * 256 + (lane & 15) * 16 + (r + hi * 8)] = acc[r];
    __syncthreads();

    for (int t = tid; t < COUT * D2; t += 256) {
        const int ch  = t / D2;
        const int ox  = t - ch * D2;
        const int mtl = ox >> 4;
        const int ml  = ox & 15;
        const int o   = ch * 16 + ml;
        const float v = pLDS[(mtl + 0) * 256 + o] + pLDS[(mtl + 2) * 256 + o]
                      + pLDS[(mtl + 4) * 256 + o] + pLDS[(mtl + 6) * 256 + o];
        // y scratch in channel-major [ch][b][voxel] for BN reduction
        y[((size_t)ch * BATCH + b) * NVOX2 + (size_t)oz * (D2 * D2) + (size_t)oy * D2 + ox] = v;
    }
}

// =====================================================================
// Batch-norm statistics: one block per channel, double accumulation.
// stats[0..15] = mean, stats[16..31] = rsqrt(var + eps)
// =====================================================================
__global__ __launch_bounds__(256) void bn_stats(const float* __restrict__ y,
                                                float* __restrict__ stats)
{
    __shared__ double red[512];
    const int ch  = blockIdx.x;
    const int tid = threadIdx.x;
    double s1 = 0.0, s2 = 0.0;
    const float* p = y + (size_t)ch * BATCH * NVOX2;
    for (int i = tid; i < BATCH * NVOX2; i += 256) {
        double v = (double)p[i];
        s1 += v; s2 += v * v;
    }
    red[tid] = s1; red[256 + tid] = s2;
    __syncthreads();
    for (int st = 128; st > 0; st >>= 1) {
        if (tid < st) {
            red[tid]       += red[tid + st];
            red[256 + tid] += red[256 + tid + st];
        }
        __syncthreads();
    }
    if (tid == 0) {
        const double n    = (double)(BATCH * NVOX2);
        const double mean = red[0] / n;
        const double var  = red[256] / n - mean * mean;
        stats[ch]        = (float)mean;
        stats[COUT + ch] = rsqrtf((float)var + 1e-5f);
    }
}

// =====================================================================
// normalize + gamma/beta + bias + relu, write NCDHW output
// =====================================================================
__global__ void bn_apply(const float* __restrict__ y,
                         const float* __restrict__ stats,
                         const float* __restrict__ gamma,
                         const float* __restrict__ beta,
                         const float* __restrict__ bias,
                         float* __restrict__ out)
{
    const int N = BATCH * COUT * NVOX2;
    int idx = blockIdx.x * blockDim.x + threadIdx.x;
    if (idx >= N) return;
    const int b   = idx / (COUT * NVOX2);
    const int r   = idx % (COUT * NVOX2);
    const int ch  = r / NVOX2;
    const int vox = r % NVOX2;
    const float v = y[((size_t)ch * BATCH + b) * NVOX2 + vox];
    const float o = (v - stats[ch]) * stats[COUT + ch] * gamma[ch] + beta[ch] + bias[ch];
    out[idx] = fmaxf(o, 0.0f);
}

// =====================================================================
extern "C" void kernel_launch(void* const* d_in, const int* in_sizes, int n_in,
                              void* d_out, int out_size, void* d_ws, size_t ws_size,
                              hipStream_t stream)
{
    const float* s       = (const float*)d_in[0];
    const float* basis1  = (const float*)d_in[1];
    const float* W1      = (const float*)d_in[2];
    const float* basis2a = (const float*)d_in[3];
    const float* basis2b = (const float*)d_in[4];
    const float* W2a     = (const float*)d_in[5];
    const float* W2b     = (const float*)d_in[6];
    const float* gamma   = (const float*)d_in[7];
    const float* beta    = (const float*)d_in[8];
    const float* bias    = (const float*)d_in[9];

    char*   ws    = (char*)d_ws;
    __bf16* K1B   = (__bf16*)(ws + OFF_K1B);
    __bf16* K2B   = (__bf16*)(ws + OFF_K2B);
    __bf16* vt    = (__bf16*)(ws + OFF_VT);
    float*  ybuf  = (float*)(ws + OFF_Y);
    float*  stats = (float*)(ws + OFF_STATS);
    float*  out   = (float*)d_out;

    build_k1<<<(K1ROWS * 32 + 255) / 256, 256, 0, stream>>>(W1, basis1, K1B);
    build_k2<<<(K2ROWS * COUT + 255) / 256, 256, 0, stream>>>(W2a, W2b, basis2a, basis2b, K2B);
    conv1_tp_kernel<<<BATCH * D1 * D1, 192, 0, stream>>>(s, K1B, vt);
    conv2_kernel<<<BATCH * D2 * D2, 256, 0, stream>>>(vt, K2B, ybuf);
    bn_stats<<<COUT, 256, 0, stream>>>(ybuf, stats);
    bn_apply<<<(BATCH * COUT * NVOX2 + 255) / 256, 256, 0, stream>>>(ybuf, stats, gamma, beta, bias, out);
}
